// EdgeGate_31112743092523
// MI455X (gfx1250) — matrix operand order repaired and checked by reading, not compiled, not run on hardware
//
#include <hip/hip_runtime.h>
#include <math.h>

typedef _Float16 f16;
typedef _Float16 v16h __attribute__((ext_vector_type(16)));
typedef float    v8f  __attribute__((ext_vector_type(8)));

#define D_NODE 128
#define D_BIAS 64
#define D_EDGE 64
#define D_HID  192
#define TILE   16
#define WAVES  8

union Frag { uint4 q[2]; v16h v; };

// ---------------------------------------------------------------------------
// Kernel 0: convert + transpose trunk weights to f16  (WT[n][k] = W[k][n])
// ---------------------------------------------------------------------------
__global__ void prep_weights(const float* __restrict__ W1,
                             const float* __restrict__ W2,
                             const float* __restrict__ Wo,
                             f16* __restrict__ wt1, f16* __restrict__ wt2,
                             f16* __restrict__ wto) {
    int i = blockIdx.x * blockDim.x + threadIdx.x;
    int stride = gridDim.x * blockDim.x;
    for (int idx = i; idx < D_HID * D_HID; idx += stride) {
        int n = idx / D_HID, k = idx % D_HID;
        wt1[idx] = (f16)W1[k * D_HID + n];
        wt2[idx] = (f16)W2[k * D_HID + n];
    }
    for (int idx = i; idx < D_EDGE * D_HID; idx += stride) {
        int n = idx / D_HID, k = idx % D_HID;
        wto[idx] = (f16)Wo[k * D_EDGE + n];
    }
}

// ---------------------------------------------------------------------------
// Kernel 1: node projection  ne = node_embed @ W_init + b_init  (f16 out)
// ---------------------------------------------------------------------------
__global__ void node_proj(const float* __restrict__ node_embed,
                          const float* __restrict__ Wi,
                          const float* __restrict__ bi,
                          f16* __restrict__ ne, int nNodes) {
    int idx = blockIdx.x * blockDim.x + threadIdx.x;
    if (idx >= nNodes * D_BIAS) return;
    int node = idx >> 6, o = idx & 63;
    const float* x = node_embed + (long)node * D_NODE;
    float acc = bi[o];
#pragma unroll 4
    for (int k = 0; k < D_NODE; k++) acc += x[k] * Wi[k * D_BIAS + o];
    ne[idx] = (f16)acc;
}

// ---------------------------------------------------------------------------
// Fragment loaders (CDNA5 wave32 16-bit WMMA layouts)
// A 16x32:  lane(m,g): K = k0+8g..k0+8g+7  and  k0+16+8g..+7   (2 x 16B)
// B 32x16:  lane(m,g): col n0+m, K = k0+8g..k0+8g+7  and  k0+16+8g..+7   (2 x 16B)
// ---------------------------------------------------------------------------
__device__ inline v16h ldA(const f16* base, int m, int k0, int g) {
    Frag f;
    f.q[0] = *(const uint4*)(base + m * D_HID + k0 + 8 * g);
    f.q[1] = *(const uint4*)(base + m * D_HID + k0 + 16 + 8 * g);
    return f.v;
}
__device__ inline v16h ldB(const f16* wt, int n0, int m, int k0, int g) {
    Frag f;
    f.q[0] = *(const uint4*)(wt + (n0 + m) * D_HID + k0 + 8 * g);
    f.q[1] = *(const uint4*)(wt + (n0 + m) * D_HID + k0 + 16 + 8 * g);
    return f.v;
}

// One 16x16 output tile: D = A(16x192) x WT(n0:n0+16, 192)^T + bias
// Two independent accumulator chains (even/odd k) for WMMA ILP.
__device__ inline v8f gemm_tile(const v16h a[6], const f16* WT, int n0, int m,
                                int g, float bias) {
    v8f acc0, acc1;
#pragma unroll
    for (int r = 0; r < 8; r++) { acc0[r] = bias; acc1[r] = 0.f; }
#pragma unroll
    for (int k = 0; k < 3; k++) {
        v16h be = ldB(WT, n0, m, (2 * k) * 32, g);
        acc0 = __builtin_amdgcn_wmma_f32_16x16x32_f16(
            false, a[2 * k], false, be, (short)0, acc0, false, false);
        v16h bo = ldB(WT, n0, m, (2 * k + 1) * 32, g);
        acc1 = __builtin_amdgcn_wmma_f32_16x16x32_f16(
            false, a[2 * k + 1], false, bo, (short)0, acc1, false, false);
    }
#pragma unroll
    for (int r = 0; r < 8; r++) acc0[r] += acc1[r];
    return acc0;
}

// ---------------------------------------------------------------------------
// Kernel 2: per-edge trunk MLP on WMMA.  One wave == one 16-edge tile.
// ---------------------------------------------------------------------------
__global__ void __launch_bounds__(256) edge_trunk(
    const float* __restrict__ edge_embed, const int* __restrict__ edge_index,
    const f16* __restrict__ ne,
    const f16* __restrict__ wt1g, const f16* __restrict__ wt2g,
    const f16* __restrict__ wtog,
    const float* __restrict__ b1, const float* __restrict__ b2,
    const float* __restrict__ bo,
    float* __restrict__ out, int E) {
    extern __shared__ char smemraw[];
    f16* WT1 = (f16*)smemraw;                 // [192][192]
    f16* WT2 = WT1 + D_HID * D_HID;           // [192][192]
    f16* WTo = WT2 + D_HID * D_HID;           // [64][192]
    f16* tiles = WTo + D_EDGE * D_HID;        // 8 waves x (inT + hT)

    const int tid = threadIdx.x;
    const int wave = tid >> 5, lane = tid & 31;
    f16* inT = tiles + wave * (2 * TILE * D_HID);
    f16* hT = inT + TILE * D_HID;

    // ---- stage all weights into LDS (f16, pre-transposed) -----------------
    {
        uint4* d1 = (uint4*)WT1; const uint4* s1 = (const uint4*)wt1g;
        for (int i = tid; i < D_HID * D_HID / 8; i += 256) d1[i] = s1[i];
        uint4* d2 = (uint4*)WT2; const uint4* s2 = (const uint4*)wt2g;
        for (int i = tid; i < D_HID * D_HID / 8; i += 256) d2[i] = s2[i];
        uint4* d3 = (uint4*)WTo; const uint4* s3 = (const uint4*)wtog;
        for (int i = tid; i < D_EDGE * D_HID / 8; i += 256) d3[i] = s3[i];
    }
    __syncthreads();

    const int m = lane & 15, g = lane >> 4;
    const int nTiles = (E + TILE - 1) / TILE;
    const int tileStride = gridDim.x * WAVES;

    // ---- hoist all biases into registers (loop-invariant) -----------------
    float b1v[12], b2v[12], bov[4];
#pragma unroll
    for (int nt = 0; nt < 12; ++nt) {
        b1v[nt] = b1[nt * 16 + m];
        b2v[nt] = b2[nt * 16 + m];
    }
#pragma unroll
    for (int nt = 0; nt < 4; ++nt) bov[nt] = bo[nt * 16 + m];

    for (int tile = blockIdx.x * WAVES + wave; tile < nTiles;
         tile += tileStride) {
        const long e0 = (long)tile * TILE;

        // ---- prefetch next tile's edge data (global_prefetch_b8) ----------
        {
            int tileN = tile + tileStride;            // wave-uniform condition
            if (tileN < nTiles) {
                const char* pe =
                    (const char*)(edge_embed + (long)tileN * TILE * D_EDGE);
                __builtin_prefetch(pe + lane * 128, 0, 3);  // 32 lanes -> 4 KB
                __builtin_prefetch(
                    (const char*)(edge_index + (long)tileN * TILE), 0, 3);
                __builtin_prefetch(
                    (const char*)(edge_index + (long)E + (long)tileN * TILE),
                    0, 3);
            }
        }

        // ---- build trunk input tile [16][192] in LDS ----------------------
        // cols 0..63: edge_embed (contiguous, coalesced across the wave)
        {
            const float4* ee = (const float4*)(edge_embed + e0 * D_EDGE);
            for (int i = lane; i < TILE * D_EDGE / 4; i += 32) {
                long erow = e0 + (i >> 4);
                float4 v = (erow < E) ? ee[i] : make_float4(0.f, 0.f, 0.f, 0.f);
                f16* d = inT + (i >> 4) * D_HID + (i & 15) * 4;
                d[0] = (f16)v.x; d[1] = (f16)v.y; d[2] = (f16)v.z; d[3] = (f16)v.w;
            }
        }
        // cols 64..127: ne[dst], cols 128..191: ne[src] (L2-resident gather)
        {
            int er = lane >> 1, part = lane & 1;   // 2 lanes per edge row
            long e = e0 + er; if (e >= E) e = E - 1;
            int dsti = edge_index[e];
            int srci = edge_index[(long)E + e];
            const uint4* pd = (const uint4*)(ne + (long)dsti * D_BIAS + part * 32);
            const uint4* ps = (const uint4*)(ne + (long)srci * D_BIAS + part * 32);
            uint4* od = (uint4*)(inT + er * D_HID + 64 + part * 32);
            uint4* os = (uint4*)(inT + er * D_HID + 128 + part * 32);
            od[0] = pd[0]; od[1] = pd[1]; od[2] = pd[2]; od[3] = pd[3];
            os[0] = ps[0]; os[1] = ps[1]; os[2] = ps[2]; os[3] = ps[3];
        }

        // ---- layer 1: hT = relu(inT @ W1 + b1) ----------------------------
        {
            v16h a[6];
#pragma unroll
            for (int k = 0; k < 6; k++) a[k] = ldA(inT, m, k * 32, g);
            for (int nt = 0; nt < 12; ++nt) {
                int n0 = nt * 16;
                v8f acc = gemm_tile(a, WT1, n0, m, g, b1v[nt]);
#pragma unroll
                for (int r = 0; r < 8; r++) {
                    float v = acc[r] > 0.f ? acc[r] : 0.f;
                    hT[(8 * g + r) * D_HID + n0 + m] = (f16)v;
                }
            }
        }

        // ---- layer 2: inT = relu(hT @ W2 + b2) ----------------------------
        {
            v16h a[6];
#pragma unroll
            for (int k = 0; k < 6; k++) a[k] = ldA(hT, m, k * 32, g);
            for (int nt = 0; nt < 12; ++nt) {
                int n0 = nt * 16;
                v8f acc = gemm_tile(a, WT2, n0, m, g, b2v[nt]);
#pragma unroll
                for (int r = 0; r < 8; r++) {
                    float v = acc[r] > 0.f ? acc[r] : 0.f;
                    inT[(8 * g + r) * D_HID + n0 + m] = (f16)v;
                }
            }
        }

        // ---- layer 3: out = sigmoid(inT @ W_out + b_out) * edge_embed -----
        {
            v16h a[6];
#pragma unroll
            for (int k = 0; k < 6; k++) a[k] = ldA(inT, m, k * 32, g);
#pragma unroll
            for (int nt = 0; nt < 4; ++nt) {
                int n0 = nt * 16;
                v8f acc = gemm_tile(a, WTo, n0, m, g, bov[nt]);
#pragma unroll
                for (int r = 0; r < 8; r++) {
                    long e = e0 + 8 * g + r;
                    if (e < E) {
                        float s = 1.f / (1.f + __expf(-acc[r]));
                        long o = e * D_EDGE + n0 + m;
                        out[o] = s * edge_embed[o];
                    }
                }
            }
        }
    }
}

// ---------------------------------------------------------------------------
extern "C" void kernel_launch(void* const* d_in, const int* in_sizes, int n_in,
                              void* d_out, int out_size, void* d_ws,
                              size_t ws_size, hipStream_t stream) {
    const float* node_embed = (const float*)d_in[0];
    const float* edge_embed = (const float*)d_in[1];
    const int*   edge_index = (const int*)d_in[2];
    const float* W_init     = (const float*)d_in[3];
    const float* b_init     = (const float*)d_in[4];
    const float* W1         = (const float*)d_in[5];
    const float* b1         = (const float*)d_in[6];
    const float* W2         = (const float*)d_in[7];
    const float* b2         = (const float*)d_in[8];
    const float* W_out      = (const float*)d_in[9];
    const float* b_out      = (const float*)d_in[10];
    float* out = (float*)d_out;

    const int nNodes = in_sizes[0] / D_NODE;
    const int E      = in_sizes[2] / 2;

    // workspace layout (f16): ne[nNodes][64] | WT1 | WT2 | WTo
    f16* ws  = (f16*)d_ws;
    f16* ne  = ws;
    size_t neCount = (((size_t)nNodes * D_BIAS) + 7) & ~(size_t)7;
    f16* wt1 = ne + neCount;
    f16* wt2 = wt1 + D_HID * D_HID;
    f16* wto = wt2 + D_HID * D_HID;

    prep_weights<<<96, 256, 0, stream>>>(W1, W2, W_out, wt1, wt2, wto);

    int total = nNodes * D_BIAS;
    node_proj<<<(total + 255) / 256, 256, 0, stream>>>(node_embed, W_init,
                                                       b_init, ne, nNodes);

    size_t smem = (size_t)(2 * D_HID * D_HID + D_EDGE * D_HID +
                           WAVES * 2 * TILE * D_HID) * sizeof(f16); // 270336 B
    (void)hipFuncSetAttribute((const void*)edge_trunk,
                              hipFuncAttributeMaxDynamicSharedMemorySize,
                              (int)smem);

    int nTiles = (E + TILE - 1) / TILE;
    int grid = 2048;
    if ((long)grid * WAVES > nTiles) grid = (nTiles + WAVES - 1) / WAVES;
    edge_trunk<<<grid, 256, smem, stream>>>(edge_embed, edge_index, ne, wt1,
                                            wt2, wto, b1, b2, b_out, out, E);
}
